// SlotGRU_13898514170409
// MI455X (gfx1250) — matrix operand order repaired and checked by reading, not compiled, not run on hardware
//
#include <hip/hip_runtime.h>
#include <stdint.h>

#define T_STEPS 256
#define BN      512      // B*N
#define F_DIM   128
#define H_DIM   256
#define G_DIM   768      // 3*H
#define NT_G    48       // G_DIM/16 column tiles
#define KT_H    8        // H_DIM/32 k-slices
#define KT_F    4        // F_DIM/32 k-slices

typedef __attribute__((ext_vector_type(16))) __bf16        v16bf;
typedef __attribute__((ext_vector_type(8)))  float         v8f;
typedef __attribute__((ext_vector_type(8)))  unsigned int  v8u;

static __device__ __forceinline__ unsigned short f2bf(float f) {
    unsigned u = __float_as_uint(f);
    unsigned r = (u + 0x7FFFu + ((u >> 16) & 1u)) >> 16;  // round-to-nearest-even
    return (unsigned short)r;
}

static __device__ __forceinline__ v8u pack_u(uint4 a, uint4 b) {
    v8u u;
    u[0] = a.x; u[1] = a.y; u[2] = a.z; u[3] = a.w;
    u[4] = b.x; u[5] = b.y; u[6] = b.z; u[7] = b.w;
    return u;
}
static __device__ __forceinline__ v16bf pack_frag(uint4 a, uint4 b) {
    return __builtin_bit_cast(v16bf, pack_u(a, b));
}

// ---------------------------------------------------------------------------
// Kernel 1: convert W_ih / W_hh (fp32, [3H,K]) into bf16 WMMA B-fragment
// blocked layout: frag (kt,nt) = 32 lanes x 16 bf16 (1 KB);
// lane -> column N = lane&15, K-half = lane>>4; element e -> K = e + 16*half.
// ---------------------------------------------------------------------------
__global__ void convert_weights_kernel(const float* __restrict__ W_ih,
                                       const float* __restrict__ W_hh,
                                       unsigned short* __restrict__ wih_b,
                                       unsigned short* __restrict__ whh_b) {
    const int NIH = NT_G * KT_F * 32 * 16;   // 98304 bf16 elements
    const int NHH = NT_G * KT_H * 32 * 16;   // 196608 bf16 elements
    int i = blockIdx.x * blockDim.x + threadIdx.x;
    if (i < NIH) {
        int e = i & 15, lane = (i >> 4) & 31, frag = i >> 9;
        int nt = frag % NT_G, kt = frag / NT_G;
        int n = nt * 16 + (lane & 15);
        int k = kt * 32 + e + 16 * (lane >> 4);
        wih_b[i] = f2bf(W_ih[n * F_DIM + k]);
    } else if (i < NIH + NHH) {
        int j = i - NIH;
        int e = j & 15, lane = (j >> 4) & 31, frag = j >> 9;
        int nt = frag % NT_G, kt = frag / NT_G;
        int n = nt * 16 + (lane & 15);
        int k = kt * 32 + e + 16 * (lane >> 4);
        whh_b[j] = f2bf(W_hh[n * H_DIM + k]);
    }
}

// ---------------------------------------------------------------------------
// Kernel 2: xg = x @ W_ih^T + b_ih, stored PRE-SWIZZLED for the scan kernel:
//   xg_s[((mtile*512 + wv*32 + lane) * 24) + g*8 + r]
// so each scan lane reads its 24 floats (8 rows x 3 gates) as 6 x b128.
// One block = one 16-row M-tile (mtile = t*32 + rowTile); 8 waves x 6 N-tiles.
// ---------------------------------------------------------------------------
__global__ void __launch_bounds__(256)
xgemm_kernel(const float* __restrict__ x,
             const unsigned short* __restrict__ wih_b,
             const float* __restrict__ b_ih,
             float* __restrict__ xg_s) {
    __shared__ __align__(16) unsigned short xbf[16][F_DIM + 8];

    const int mtile = blockIdx.x;
    const long rowBase = (long)mtile * 16;
    const int tid = threadIdx.x;

    // stage x tile -> bf16 LDS
    for (int idx = tid; idx < 16 * F_DIM; idx += 256) {
        int rr = idx >> 7, k = idx & 127;
        xbf[rr][k] = f2bf(x[(rowBase + rr) * F_DIM + k]);
    }
    __syncthreads();

    const int lane = tid & 31;
    const int wave = tid >> 5;          // 0..7
    const int half = lane >> 4;
    const int m    = lane & 15;

    for (int nti = 0; nti < 6; ++nti) {
        const int nt = wave * 6 + nti;      // 0..47
        const int g  = nt >> 4;             // gate block 0..2
        const int wv = nt & 15;             // scan-wave that consumes this tile
        v8f acc = {};
        for (int kt = 0; kt < KT_F; ++kt) {
            const int kb = kt * 32 + 16 * half;
            uint4 a0 = *(const uint4*)&xbf[m][kb];       // K = kb..kb+7
            uint4 a1 = *(const uint4*)&xbf[m][kb + 8];   // K = kb+8..kb+15
            const uint4* bp = (const uint4*)(wih_b +
                              (size_t)(kt * NT_G + nt) * 512 + lane * 16);
            uint4 b0 = bp[0], b1 = bp[1];
            acc = __builtin_amdgcn_wmma_f32_16x16x32_bf16(
                false, pack_frag(a0, a1), false, pack_frag(b0, b1),
                (short)0, acc, false, false);
        }
        const float bi = b_ih[nt * 16 + m];
        float* dst = xg_s + ((size_t)mtile * 512 + (size_t)wv * 32 + lane) * 24 + g * 8;
        float4 s0, s1;
        s0.x = acc[0] + bi; s0.y = acc[1] + bi; s0.z = acc[2] + bi; s0.w = acc[3] + bi;
        s1.x = acc[4] + bi; s1.y = acc[5] + bi; s1.z = acc[6] + bi; s1.w = acc[7] + bi;
        *(float4*)dst       = s0;
        *(float4*)(dst + 4) = s1;
    }
}

// ---------------------------------------------------------------------------
// Kernel 3: persistent GRU scan. Block = 16 rows of BN, all 256 steps local.
// 512 threads = 16 waves. Wave w owns gate tiles {w, 16+w, 32+w}, i.e. the
// r/z/n tiles for H-columns [16w, 16w+16): gate combine is wave-local, in
// registers. W_hh fragments are loop-invariant -> held resident in VGPRs.
// Per step: 2 barriers, 16 ds_read_b128 (A), 24 WMMA, 6+2 b128 loads, 8 stores.
// ---------------------------------------------------------------------------
__global__ void __launch_bounds__(512)
gru_scan_kernel(const float* __restrict__ xg_s,
                const int*   __restrict__ dones,
                const float* __restrict__ h0,
                const unsigned short* __restrict__ whh_b,
                const float* __restrict__ b_hh,
                float* __restrict__ out) {
    __shared__ __align__(16) unsigned short hbf[16][H_DIM + 8];   // 8.4 KB

    const int tid = threadIdx.x;
    const int rowTile = blockIdx.x;
    const int rowBase = rowTile * 16;
    const int lane = tid & 31;
    const int w    = tid >> 5;          // 0..15
    const int half = lane >> 4;
    const int m    = lane & 15;
    const int col  = w * 16 + m;        // H column this lane owns (C layout N)
    const int rb   = 8 * half;          // first owned row (C layout M base)

    // loop-invariant biases for this lane's column
    const float b_r = b_hh[col];
    const float b_z = b_hh[H_DIM + col];
    const float b_n = b_hh[2 * H_DIM + col];

    // resident W_hh B-fragments: 3 gates x 8 k-slices x 8 VGPRs = 192 VGPRs
    v8u Br[KT_H], Bz[KT_H], Bn[KT_H];
#pragma unroll
    for (int kt = 0; kt < KT_H; ++kt) {
        const uint4* pr = (const uint4*)(whh_b + (size_t)(kt * NT_G +      w) * 512 + lane * 16);
        const uint4* pz = (const uint4*)(whh_b + (size_t)(kt * NT_G + 16 + w) * 512 + lane * 16);
        const uint4* pn = (const uint4*)(whh_b + (size_t)(kt * NT_G + 32 + w) * 512 + lane * 16);
        Br[kt] = pack_u(pr[0], pr[1]);
        Bz[kt] = pack_u(pz[0], pz[1]);
        Bn[kt] = pack_u(pn[0], pn[1]);
    }

    // register-resident h: rows rb..rb+7, column col  (matches WMMA C layout)
    float h[8];
#pragma unroll
    for (int i = 0; i < 8; ++i)
        h[i] = h0[(size_t)(rowBase + rb + i) * H_DIM + col];

    for (int t = 0; t < T_STEPS; ++t) {
        // done-mask reset, publish bf16 h to LDS for A-fragments
        const int* dp = dones + (size_t)t * BN + rowBase + rb;
        uint4 d0 = *(const uint4*)dp;
        uint4 d1 = *(const uint4*)(dp + 4);
        int di[8] = {(int)d0.x, (int)d0.y, (int)d0.z, (int)d0.w,
                     (int)d1.x, (int)d1.y, (int)d1.z, (int)d1.w};
#pragma unroll
        for (int i = 0; i < 8; ++i) {
            h[i] *= 1.0f - (float)di[i];
            hbf[rb + i][col] = f2bf(h[i]);
        }
        __syncthreads();

        // hg = h @ W_hh^T for this wave's column triple, all in registers
        v8f ar = {}, az = {}, an = {};
#pragma unroll
        for (int kt = 0; kt < KT_H; ++kt) {
            const int kb = kt * 32 + 16 * half;
            uint4 a0 = *(const uint4*)&hbf[m][kb];       // K = kb..kb+7
            uint4 a1 = *(const uint4*)&hbf[m][kb + 8];   // K = kb+8..kb+15
            v16bf A = pack_frag(a0, a1);
            ar = __builtin_amdgcn_wmma_f32_16x16x32_bf16(
                false, A, false, __builtin_bit_cast(v16bf, Br[kt]),
                (short)0, ar, false, false);
            az = __builtin_amdgcn_wmma_f32_16x16x32_bf16(
                false, A, false, __builtin_bit_cast(v16bf, Bz[kt]),
                (short)0, az, false, false);
            an = __builtin_amdgcn_wmma_f32_16x16x32_bf16(
                false, A, false, __builtin_bit_cast(v16bf, Bn[kt]),
                (short)0, an, false, false);
        }

        // this lane's 24 xg values: 6 x b128
        const float* xp = xg_s +
            ((size_t)((size_t)t * 32 + rowTile) * 512 + tid) * 24;
        const float4* xv = (const float4*)xp;
        float4 X0 = xv[0], X1 = xv[1];   // r gate, rows 0..7
        float4 X2 = xv[2], X3 = xv[3];   // z gate
        float4 X4 = xv[4], X5 = xv[5];   // n gate
        float Xr[8] = {X0.x, X0.y, X0.z, X0.w, X1.x, X1.y, X1.z, X1.w};
        float Xz[8] = {X2.x, X2.y, X2.z, X2.w, X3.x, X3.y, X3.z, X3.w};
        float Xn[8] = {X4.x, X4.y, X4.z, X4.w, X5.x, X5.y, X5.z, X5.w};

        // prefetch next step's xg block while we do the gate math
        if (t + 1 < T_STEPS)
            __builtin_prefetch(xp + (size_t)32 * 512 * 24, 0, 0);

        size_t obase = (size_t)t * BN * H_DIM + (size_t)(rowBase + rb) * H_DIM + col;
#pragma unroll
        for (int i = 0; i < 8; ++i) {
            float rg = 1.0f / (1.0f + __expf(-(ar[i] + Xr[i] + b_r)));
            float zg = 1.0f / (1.0f + __expf(-(az[i] + Xz[i] + b_z)));
            float ng = tanhf(Xn[i] + rg * (an[i] + b_n));
            h[i] = (1.0f - zg) * ng + zg * h[i];
            out[obase + (size_t)i * H_DIM] = h[i];
        }
        __syncthreads();   // all waves done reading hbf before next publish
    }

    // final_h
    size_t fbase = (size_t)T_STEPS * BN * H_DIM +
                   (size_t)(rowBase + rb) * H_DIM + col;
#pragma unroll
    for (int i = 0; i < 8; ++i)
        out[fbase + (size_t)i * H_DIM] = h[i];
}

// ---------------------------------------------------------------------------
extern "C" void kernel_launch(void* const* d_in, const int* in_sizes, int n_in,
                              void* d_out, int out_size, void* d_ws, size_t ws_size,
                              hipStream_t stream) {
    (void)in_sizes; (void)n_in; (void)out_size; (void)ws_size;

    const float* x     = (const float*)d_in[0];
    const int*   dones = (const int*)  d_in[1];
    const float* h0    = (const float*)d_in[2];
    const float* W_ih  = (const float*)d_in[3];
    const float* W_hh  = (const float*)d_in[4];
    const float* b_ih  = (const float*)d_in[5];
    const float* b_hh  = (const float*)d_in[6];
    float* out = (float*)d_out;

    char* ws = (char*)d_ws;
    unsigned short* wih_b = (unsigned short*)ws;                    // 196608 B
    unsigned short* whh_b = (unsigned short*)(ws + 196608);         // 393216 B
    float* xg_s = (float*)(ws + 196608 + 393216);                   // 384 MB

    const int NCONV = NT_G * KT_F * 32 * 16 + NT_G * KT_H * 32 * 16; // 294912
    convert_weights_kernel<<<(NCONV + 255) / 256, 256, 0, stream>>>(
        W_ih, W_hh, wih_b, whh_b);

    xgemm_kernel<<<(T_STEPS * BN) / 16, 256, 0, stream>>>(x, wih_b, b_ih, xg_s);

    gru_scan_kernel<<<BN / 16, 512, 0, stream>>>(xg_s, dones, h0, whh_b, b_hh, out);
}
